// AdvancedHybridBBBNetEnhanced_39651138076874
// MI455X (gfx1250) — matrix operand order repaired and checked by reading, not compiled, not run on hardware
//
#include <hip/hip_runtime.h>
#include <hip/hip_bf16.h>

// ---------------- problem constants (match reference) ----------------
static constexpr int kN = 50000;   // 50000 = 16 * 3125  -> all GEMM M are multiples of 16
static constexpr int kE = 800000;
static constexpr int kG = 256;
static constexpr int kF = 34;
static constexpr int kD = 128;
static constexpr int kH = 4;
static constexpr int kC = 32;
static constexpr int kL = 4;
static constexpr unsigned ENC_NEG_INF = 0x007FFFFFu; // fenc(-inf)

typedef _Float16 half_t;
typedef __attribute__((ext_vector_type(8)))  _Float16 v8h;
typedef __attribute__((ext_vector_type(16))) _Float16 v16h;
typedef __attribute__((ext_vector_type(8)))  float    v8f;

#define GRID(n, b) (((n) + (b) - 1) / (b))

// order-preserving float <-> uint encoding for atomic max
__device__ __forceinline__ unsigned fenc(float f) {
  unsigned u = __float_as_uint(f);
  return (u & 0x80000000u) ? ~u : (u | 0x80000000u);
}
__device__ __forceinline__ float fdec(unsigned u) {
  return __uint_as_float((u & 0x80000000u) ? (u & 0x7FFFFFFFu) : ~u);
}

// ---------------- fills / converts ----------------
__global__ void k_fill_f32(float* p, float v, int n) {
  int i = blockIdx.x * blockDim.x + threadIdx.x;
  if (i < n) p[i] = v;
}
__global__ void k_fill_u32(unsigned* p, unsigned v, int n) {
  int i = blockIdx.x * blockDim.x + threadIdx.x;
  if (i < n) p[i] = v;
}
__global__ void k_fill_bias_rows(float* dst, const float* __restrict__ bias, int rows, int cols) {
  int i = blockIdx.x * blockDim.x + threadIdx.x;
  if (i < rows * cols) dst[i] = bias[i % cols];
}
__global__ void k_cvt_f16(const float* __restrict__ s, half_t* __restrict__ d, int n) {
  int i = blockIdx.x * blockDim.x + threadIdx.x;
  if (i < n) d[i] = (half_t)s[i];
}
// pad columns with zeros (rows x cols -> rows x padCols), row-major A input
__global__ void k_cvt_f16_padcols(const float* __restrict__ s, half_t* __restrict__ d,
                                  int rows, int cols, int padCols) {
  int i = blockIdx.x * blockDim.x + threadIdx.x;
  if (i >= rows * padCols) return;
  int r = i / padCols, c = i % padCols;
  d[i] = (c < cols) ? (half_t)s[(size_t)r * cols + c] : (half_t)0.f;
}
// weight transpose to f16: src (K x Nc) f32 -> dst (Nc x Kpad) f16, zero-pad K
__global__ void k_cvt_f16_T(const float* __restrict__ s, half_t* __restrict__ d,
                            int Ksrc, int Kpad, int Nc) {
  int i = blockIdx.x * blockDim.x + threadIdx.x;
  if (i >= Nc * Kpad) return;
  int c = i / Kpad, k = i % Kpad;
  d[i] = (k < Ksrc) ? (half_t)s[(size_t)k * Nc + c] : (half_t)0.f;
}

// ---------------- WMMA GEMM: C[MxNc] = A[MxK](f16) * B[KxNc] + bias ----------------
// Bt is the transposed weight, (Nc x K) f16 row-major, so every lane's B fragment
// is two contiguous 16B runs, with the same K split as the A fragment below.
// A fragment per lane = two contiguous 16B runs (lanes 0-15: K 0-7 & 16-23;
// lanes 16-31: K 8-15 & 24-31). All loads lower to global_load_b128.
// M is a multiple of 16 for every call site -> unguarded stores.
template <int K>
__global__ void k_gemm_wmma(const half_t* __restrict__ A, const half_t* __restrict__ Bt,
                            const float* __restrict__ bias, float* __restrict__ Cout,
                            int M, int Nc, int relu) {
  const int lane = threadIdx.x & 31;
  const int wave = blockIdx.x * (blockDim.x >> 5) + (threadIdx.x >> 5);
  const int Nt = Nc >> 4;
  const int tileM = wave / Nt;
  const int tileN = wave % Nt;
  if (tileM * 16 >= M) return;              // wave-uniform exit: EXEC stays all-1s for WMMA
  const int lo = lane & 15, hi = lane >> 4;

  const half_t* arow = A + (size_t)(tileM * 16 + lo) * K;
  const half_t* brow = Bt + (size_t)(tileN * 16 + lo) * K;

  v8f acc;
  const float bv = bias[tileN * 16 + lo];
#pragma unroll
  for (int r = 0; r < 8; ++r) acc[r] = bv;

#pragma unroll
  for (int k0 = 0; k0 < K; k0 += 32) {
    const v8h* pa = (const v8h*)(arow + k0);
    v8h a0 = pa[hi];            // K = k0 + hi*8 .. +7
    v8h a1 = pa[2 + hi];        // K = k0 + 16 + hi*8 .. +7
    const v8h* pb = (const v8h*)(brow + k0);
    v8h b0 = pb[hi];            // K = k0 + hi*8 .. +7
    v8h b1 = pb[2 + hi];        // K = k0 + 16 + hi*8 .. +7
    v16h a = __builtin_shufflevector(a0, a1, 0, 1, 2, 3, 4, 5, 6, 7,
                                     8, 9, 10, 11, 12, 13, 14, 15);
    v16h b = __builtin_shufflevector(b0, b1, 0, 1, 2, 3, 4, 5, 6, 7,
                                     8, 9, 10, 11, 12, 13, 14, 15);
    acc = __builtin_amdgcn_wmma_f32_16x16x32_f16(false, a, false, b, (short)0, acc, false, false);
  }

  float* crow = Cout + (size_t)(tileM * 16 + hi * 8) * Nc + tileN * 16 + lo;
#pragma unroll
  for (int r = 0; r < 8; ++r) {  // C/D: VGPR r -> M=r (lanes 0-15), M=r+8 (lanes 16-31)
    float v = acc[r];
    if (relu) v = fmaxf(v, 0.f);
    crow[(size_t)r * Nc] = v;
  }
}

// ---------------- BatchNorm (train-mode batch statistics) ----------------
__global__ void k_col_stats(const float* __restrict__ x, int M, int Nc,
                            float* __restrict__ mean, float* __restrict__ var) {
  __shared__ float s1[256], s2[256];
  const int col = blockIdx.x;
  float a = 0.f, b = 0.f;
  for (int r = threadIdx.x; r < M; r += blockDim.x) {
    float v = x[(size_t)r * Nc + col];
    a += v; b += v * v;
  }
  s1[threadIdx.x] = a; s2[threadIdx.x] = b;
  __syncthreads();
  for (int s = 128; s > 0; s >>= 1) {
    if ((int)threadIdx.x < s) { s1[threadIdx.x] += s1[threadIdx.x + s]; s2[threadIdx.x] += s2[threadIdx.x + s]; }
    __syncthreads();
  }
  if (threadIdx.x == 0) {
    float m = s1[0] / (float)M;
    mean[col] = m;
    var[col] = s2[0] / (float)M - m * m;
  }
}
__global__ void k_bn_apply(const float* __restrict__ x, const float* __restrict__ mean,
                           const float* __restrict__ var, const float* __restrict__ g,
                           const float* __restrict__ b, const float* __restrict__ residual,
                           float* __restrict__ y, int M, int Nc, int relu) {
  int i = blockIdx.x * blockDim.x + threadIdx.x;
  if (i >= M * Nc) return;
  int c = i % Nc;
  float v = g[c] * (x[i] - mean[c]) * rsqrtf(var[c] + 1e-5f) + b[c];
  if (relu) v = fmaxf(v, 0.f);
  if (residual) v += residual[i];
  y[i] = v;
}

// ---------------- GATv2 edge kernels (edges include appended self-loops) ----------------
__global__ void k_gat_edge_score(const float* __restrict__ xl, const float* __restrict__ xr,
                                 const int* __restrict__ ei, int E_,
                                 const float* __restrict__ att, float* __restrict__ sc,
                                 unsigned* __restrict__ menc, int total) {
  int idx = blockIdx.x * blockDim.x + threadIdx.x;
  if (idx >= total) return;
  int e = idx / kH, h = idx % kH;
  int s, d;
  if (e < E_) { s = ei[e]; d = ei[E_ + e]; } else { s = d = e - E_; }
  const float* pl = xl + (size_t)s * kD + h * kC;
  const float* pr = xr + (size_t)d * kD + h * kC;
  const float* pa = att + h * kC;
  float acc = 0.f;
#pragma unroll
  for (int c = 0; c < kC; ++c) {
    float v = pl[c] + pr[c];
    v = (v > 0.f) ? v : 0.2f * v;  // leaky_relu(0.2)
    acc += v * pa[c];
  }
  sc[idx] = acc;
  atomicMax(menc + (size_t)d * kH + h, fenc(acc));
}

__global__ void k_edge_exp(float* __restrict__ sc, const unsigned* __restrict__ menc,
                           float* __restrict__ den, const int* __restrict__ ei,
                           int E_, int total) {
  int idx = blockIdx.x * blockDim.x + threadIdx.x;
  if (idx >= total) return;
  int e = idx / kH, h = idx % kH;
  int d = (e < E_) ? ei[E_ + e] : (e - E_);
  float m = fdec(menc[(size_t)d * kH + h]);
  if (!isfinite(m)) m = 0.f;                 // reference guard for empty segments
  float ex = __expf(sc[idx] - m);
  sc[idx] = ex;
  atomicAdd(den + (size_t)d * kH + h, ex);
}

__global__ void k_edge_agg(const float* __restrict__ msg, const float* __restrict__ sc,
                           const float* __restrict__ den, const int* __restrict__ ei,
                           int E_, float* __restrict__ out, int total) {
  int idx = blockIdx.x * blockDim.x + threadIdx.x;
  if (idx >= total) return;
  int e = idx / kH, h = idx % kH;
  int s, d;
  if (e < E_) { s = ei[e]; d = ei[E_ + e]; } else { s = d = e - E_; }
  float a = sc[idx] / (den[(size_t)d * kH + h] + 1e-16f);
  const float* pm = msg + (size_t)s * kD + h * kC;
  float* po = out + (size_t)d * kD + h * kC;
#pragma unroll
  for (int c = 0; c < kC; ++c) atomicAdd(po + c, pm[c] * a);
}

// ---------------- TransformerConv score (dot-product attention) ----------------
__global__ void k_tc_edge_score(const float* __restrict__ q, const float* __restrict__ kk,
                                const int* __restrict__ ei, int E_, float* __restrict__ sc,
                                unsigned* __restrict__ menc, int total) {
  int idx = blockIdx.x * blockDim.x + threadIdx.x;
  if (idx >= total) return;
  int e = idx / kH, h = idx % kH;
  int s = ei[e], d = ei[E_ + e];
  const float* pq = q + (size_t)d * kD + h * kC;
  const float* pk = kk + (size_t)s * kD + h * kC;
  float acc = 0.f;
#pragma unroll
  for (int c = 0; c < kC; ++c) acc += pq[c] * pk[c];
  acc *= 0.17677669529663687f;  // 1/sqrt(32)
  sc[idx] = acc;
  atomicMax(menc + (size_t)d * kH + h, fenc(acc));
}

// ---------------- pooling ----------------
__global__ void k_pool_accum(const float* __restrict__ x, const int* __restrict__ batch,
                             float* __restrict__ sum, unsigned* __restrict__ mx,
                             float* __restrict__ cnt, int N_) {
  int i = blockIdx.x * blockDim.x + threadIdx.x;
  if (i >= N_ * kD) return;
  int n = i / kD, d = i % kD;
  int g = batch[n];
  float v = x[i];
  atomicAdd(sum + (size_t)g * kD + d, v);
  atomicMax(mx + (size_t)g * kD + d, fenc(v));
  if (d == 0) atomicAdd(cnt + g, 1.0f);
}
__global__ void k_pool_finalize(const float* __restrict__ sum, const unsigned* __restrict__ mx,
                                const float* __restrict__ cnt, float* __restrict__ h, int G_) {
  int i = blockIdx.x * blockDim.x + threadIdx.x;
  if (i >= G_ * kD) return;
  int g = i / kD, d = i % kD;
  float c = fmaxf(cnt[g], 1.0f);
  h[(size_t)g * (2 * kD) + d] = sum[i] / c;
  float m = fdec(mx[i]);
  h[(size_t)g * (2 * kD) + kD + d] = isfinite(m) ? m : 0.f;
}

// ---------------- final 64->1 projection ----------------
__global__ void k_head_final(const float* __restrict__ t2, const float* __restrict__ W3,
                             const float* __restrict__ b3, float* __restrict__ out, int G_) {
  int g = blockIdx.x * blockDim.x + threadIdx.x;
  if (g >= G_) return;
  float acc = b3[0];
#pragma unroll
  for (int i = 0; i < 64; ++i) acc += t2[(size_t)g * 64 + i] * W3[i];
  out[g] = acc;
}

// ===================================================================
extern "C" void kernel_launch(void* const* d_in, const int* in_sizes, int n_in,
                              void* d_out, int out_size, void* d_ws, size_t ws_size,
                              hipStream_t stream) {
  (void)in_sizes; (void)n_in; (void)out_size; (void)ws_size;
  const float* x_in   = (const float*)d_in[0];
  const int*   ei     = (const int*)  d_in[1];   // [2,E] row-major: src=ei[0..E), dst=ei[E..2E)
  const int*   batch  = (const int*)  d_in[2];
  const float* emb_W  = (const float*)d_in[3];
  const float* emb_b  = (const float*)d_in[4];
  const float* emb_g  = (const float*)d_in[5];
  const float* emb_be = (const float*)d_in[6];
  const float* gWl    = (const float*)d_in[7];
  const float* gWr    = (const float*)d_in[8];
  const float* gbl    = (const float*)d_in[9];
  const float* gbr    = (const float*)d_in[10];
  const float* gatt   = (const float*)d_in[11];
  const float* gbias  = (const float*)d_in[12];
  const float* gg     = (const float*)d_in[13];
  const float* gbe    = (const float*)d_in[14];
  const float* tWq    = (const float*)d_in[15];
  const float* tWk    = (const float*)d_in[16];
  const float* tWv    = (const float*)d_in[17];
  const float* tWs    = (const float*)d_in[18];
  const float* tbq    = (const float*)d_in[19];
  const float* tbk    = (const float*)d_in[20];
  const float* tbv    = (const float*)d_in[21];
  const float* tbs    = (const float*)d_in[22];
  const float* tg     = (const float*)d_in[23];
  const float* tbe    = (const float*)d_in[24];
  const float* oW1    = (const float*)d_in[25];
  const float* ob1    = (const float*)d_in[26];
  const float* og     = (const float*)d_in[27];
  const float* obe    = (const float*)d_in[28];
  const float* oW2    = (const float*)d_in[29];
  const float* ob2    = (const float*)d_in[30];
  const float* oW3    = (const float*)d_in[31];
  const float* ob3    = (const float*)d_in[32];
  float* out = (float*)d_out;

  // ---- workspace carve-out (all 256B aligned; ~170 MB total) ----
  char* p = (char*)d_ws;
  auto alloc = [&](size_t bytes) -> char* {
    char* r = p; p += (bytes + 255) & ~(size_t)255; return r;
  };
  const size_t ND = (size_t)kN * kD;
  float*    xA    = (float*)alloc(ND * 4);
  float*    xB    = (float*)alloc(ND * 4);
  float*    bufL  = (float*)alloc(ND * 4);
  float*    bufR  = (float*)alloc(ND * 4);
  float*    bufV  = (float*)alloc(ND * 4);
  float*    bufO  = (float*)alloc(ND * 4);
  half_t*   x16   = (half_t*)alloc(ND * 2);          // also hosts padded N x 64 emb input
  float*    sc    = (float*)alloc((size_t)(kE + kN) * kH * 4);
  float*    den   = (float*)alloc((size_t)kN * kH * 4);
  unsigned* menc  = (unsigned*)alloc((size_t)kN * kH * 4);
  float*    meanb = (float*)alloc(256 * 4);
  float*    varb  = (float*)alloc(256 * 4);
  // transposed f16 weights (Nc x K row-major)
  half_t*   embWT = (half_t*)alloc((size_t)kD * 64 * 2);        // 128 x 64
  half_t*   gWlT  = (half_t*)alloc((size_t)kL * kD * kD * 2);
  half_t*   gWrT  = (half_t*)alloc((size_t)kL * kD * kD * 2);
  half_t*   tWqT  = (half_t*)alloc((size_t)kD * kD * 2);
  half_t*   tWkT  = (half_t*)alloc((size_t)kD * kD * 2);
  half_t*   tWvT  = (half_t*)alloc((size_t)kD * kD * 2);
  half_t*   tWsT  = (half_t*)alloc((size_t)kD * kD * 2);
  half_t*   oW1T  = (half_t*)alloc((size_t)kD * 2 * kD * 2);    // 128 x 256
  half_t*   oW2T  = (half_t*)alloc((size_t)64 * kD * 2);        // 64 x 128
  float*    psum  = (float*)alloc((size_t)kG * kD * 4);
  unsigned* pmax  = (unsigned*)alloc((size_t)kG * kD * 4);
  float*    pcnt  = (float*)alloc((size_t)kG * 4);
  float*    hpool = (float*)alloc((size_t)kG * 2 * kD * 4);
  half_t*   h16   = (half_t*)alloc((size_t)kG * 2 * kD * 2);
  float*    t1    = (float*)alloc((size_t)kG * kD * 4);
  half_t*   t116  = (half_t*)alloc((size_t)kG * kD * 2);
  float*    t2    = (float*)alloc((size_t)kG * 64 * 4);

  const int B = 256;
  auto gemm64 = [&](const half_t* A, const half_t* W, const float* bias, float* Cp,
                    int M, int Nc, int relu) {
    int waves = (M / 16) * (Nc / 16);
    k_gemm_wmma<64><<<GRID(waves, 8), B, 0, stream>>>(A, W, bias, Cp, M, Nc, relu);
  };
  auto gemm128 = [&](const half_t* A, const half_t* W, const float* bias, float* Cp,
                     int M, int Nc, int relu) {
    int waves = (M / 16) * (Nc / 16);
    k_gemm_wmma<128><<<GRID(waves, 8), B, 0, stream>>>(A, W, bias, Cp, M, Nc, relu);
  };
  auto gemm256 = [&](const half_t* A, const half_t* W, const float* bias, float* Cp,
                     int M, int Nc, int relu) {
    int waves = (M / 16) * (Nc / 16);
    k_gemm_wmma<256><<<GRID(waves, 8), B, 0, stream>>>(A, W, bias, Cp, M, Nc, relu);
  };

  // ---- weight conversions: transpose to f16 (Nc x K), redone every launch ----
  k_cvt_f16_T<<<GRID(kD * 64, B), B, 0, stream>>>(emb_W, embWT, kF, 64, kD);
  for (int l = 0; l < kL; ++l) {
    k_cvt_f16_T<<<GRID(kD * kD, B), B, 0, stream>>>(gWl + (size_t)l * kD * kD,
                                                    gWlT + (size_t)l * kD * kD, kD, kD, kD);
    k_cvt_f16_T<<<GRID(kD * kD, B), B, 0, stream>>>(gWr + (size_t)l * kD * kD,
                                                    gWrT + (size_t)l * kD * kD, kD, kD, kD);
  }
  k_cvt_f16_T<<<GRID(kD * kD, B), B, 0, stream>>>(tWq, tWqT, kD, kD, kD);
  k_cvt_f16_T<<<GRID(kD * kD, B), B, 0, stream>>>(tWk, tWkT, kD, kD, kD);
  k_cvt_f16_T<<<GRID(kD * kD, B), B, 0, stream>>>(tWv, tWvT, kD, kD, kD);
  k_cvt_f16_T<<<GRID(kD * kD, B), B, 0, stream>>>(tWs, tWsT, kD, kD, kD);
  k_cvt_f16_T<<<GRID(kD * 2 * kD, B), B, 0, stream>>>(oW1, oW1T, 2 * kD, 2 * kD, kD);
  k_cvt_f16_T<<<GRID(64 * kD, B), B, 0, stream>>>(oW2, oW2T, kD, kD, 64);

  // ---- embedding: relu(BN(x @ emb_W + emb_b)) ----
  k_cvt_f16_padcols<<<GRID(kN * 64, B), B, 0, stream>>>(x_in, x16, kN, kF, 64);
  gemm64(x16, embWT, emb_b, bufO, kN, kD, 0);
  k_col_stats<<<kD, 256, 0, stream>>>(bufO, kN, kD, meanb, varb);
  k_bn_apply<<<GRID((int)ND, B), B, 0, stream>>>(bufO, meanb, varb, emb_g, emb_be,
                                                 nullptr, xA, kN, kD, 1);
  float* cur = xA; float* nxt = xB;

  // ---- 4 GATv2 layers ----
  const int totG = (kE + kN) * kH;
  for (int l = 0; l < kL; ++l) {
    k_cvt_f16<<<GRID((int)ND, B), B, 0, stream>>>(cur, x16, (int)ND);
    gemm128(x16, gWlT + (size_t)l * kD * kD, gbl + (size_t)l * kD, bufL, kN, kD, 0);
    gemm128(x16, gWrT + (size_t)l * kD * kD, gbr + (size_t)l * kD, bufR, kN, kD, 0);
    k_fill_u32<<<GRID(kN * kH, B), B, 0, stream>>>(menc, ENC_NEG_INF, kN * kH);
    k_fill_f32<<<GRID(kN * kH, B), B, 0, stream>>>(den, 0.f, kN * kH);
    k_fill_bias_rows<<<GRID((int)ND, B), B, 0, stream>>>(bufO, gbias + (size_t)l * kD, kN, kD);
    k_gat_edge_score<<<GRID(totG, B), B, 0, stream>>>(bufL, bufR, ei, kE,
                                                      gatt + (size_t)l * kH * kC, sc, menc, totG);
    k_edge_exp<<<GRID(totG, B), B, 0, stream>>>(sc, menc, den, ei, kE, totG);
    k_edge_agg<<<GRID(totG, B), B, 0, stream>>>(bufL, sc, den, ei, kE, bufO, totG);
    k_col_stats<<<kD, 256, 0, stream>>>(bufO, kN, kD, meanb, varb);
    k_bn_apply<<<GRID((int)ND, B), B, 0, stream>>>(bufO, meanb, varb, gg + (size_t)l * kD,
                                                   gbe + (size_t)l * kD, cur, nxt, kN, kD, 1);
    float* t = cur; cur = nxt; nxt = t;
  }

  // ---- TransformerConv ----
  const int totT = kE * kH;
  k_cvt_f16<<<GRID((int)ND, B), B, 0, stream>>>(cur, x16, (int)ND);
  gemm128(x16, tWqT, tbq, bufL, kN, kD, 0);
  gemm128(x16, tWkT, tbk, bufR, kN, kD, 0);
  gemm128(x16, tWvT, tbv, bufV, kN, kD, 0);
  gemm128(x16, tWsT, tbs, bufO, kN, kD, 0);         // root/skip term; agg adds on top
  k_fill_u32<<<GRID(kN * kH, B), B, 0, stream>>>(menc, ENC_NEG_INF, kN * kH);
  k_fill_f32<<<GRID(kN * kH, B), B, 0, stream>>>(den, 0.f, kN * kH);
  k_tc_edge_score<<<GRID(totT, B), B, 0, stream>>>(bufL, bufR, ei, kE, sc, menc, totT);
  k_edge_exp<<<GRID(totT, B), B, 0, stream>>>(sc, menc, den, ei, kE, totT);
  k_edge_agg<<<GRID(totT, B), B, 0, stream>>>(bufV, sc, den, ei, kE, bufO, totT);
  k_col_stats<<<kD, 256, 0, stream>>>(bufO, kN, kD, meanb, varb);
  k_bn_apply<<<GRID((int)ND, B), B, 0, stream>>>(bufO, meanb, varb, tg, tbe, cur, nxt, kN, kD, 0);
  { float* t = cur; cur = nxt; nxt = t; }

  // ---- mean + max pooling over graphs ----
  k_fill_f32<<<GRID(kG * kD, B), B, 0, stream>>>(psum, 0.f, kG * kD);
  k_fill_u32<<<GRID(kG * kD, B), B, 0, stream>>>(pmax, ENC_NEG_INF, kG * kD);
  k_fill_f32<<<GRID(kG, B), B, 0, stream>>>(pcnt, 0.f, kG);
  k_pool_accum<<<GRID((int)ND, B), B, 0, stream>>>(cur, batch, psum, pmax, pcnt, kN);
  k_pool_finalize<<<GRID(kG * kD, B), B, 0, stream>>>(psum, pmax, pcnt, hpool, kG);

  // ---- MLP head ----
  k_cvt_f16<<<GRID(kG * 2 * kD, B), B, 0, stream>>>(hpool, h16, kG * 2 * kD);
  gemm256(h16, oW1T, ob1, t1, kG, kD, 0);
  k_col_stats<<<kD, 256, 0, stream>>>(t1, kG, kD, meanb, varb);
  k_bn_apply<<<GRID(kG * kD, B), B, 0, stream>>>(t1, meanb, varb, og, obe, nullptr, t1, kG, kD, 1);
  k_cvt_f16<<<GRID(kG * kD, B), B, 0, stream>>>(t1, t116, kG * kD);
  gemm128(t116, oW2T, ob2, t2, kG, 64, 1);
  k_head_final<<<GRID(kG, 64), 64, 0, stream>>>(t2, oW3, ob3, out, kG);
}
